// LossContrastiveMSCRG_11433202942636
// MI455X (gfx1250) — matrix operand order repaired and checked by reading, not compiled, not run on hardware
//
#include <hip/hip_runtime.h>

// Problem constants (match reference)
#define Bq 8192
#define Lc 256
#define Fd 128
#define Tt (Bq + Lc)          // 8448 = 528 * 16
#define TEMPf 0.07f
#define EPSf  1e-8f

typedef __attribute__((ext_vector_type(16))) _Float16 v16h;
typedef __attribute__((ext_vector_type(8)))  _Float16 v8h;
typedef __attribute__((ext_vector_type(8)))  float    v8f;
typedef __attribute__((ext_vector_type(4)))  float    v4f;

// ---------------------------------------------------------------------------
// WMMA operand loaders (CDNA5 wave32 layouts, CDNA5 ISA §7.12.2)
// A 16x32 f16: lane = (m | half<<4); elems 0..7 = K[half*8 .. +7],
//              elems 8..15 = K[16+half*8 .. +7]  (two contiguous 16B chunks)
// B 32x16 f16: lane n = lane&15; elems in k as for A above  -> strided (ld) column load when the
//              source matrix is stored K-major (i.e. transposed copies)
// C/D 16x16 f32: vgpr r, lane ln -> (m = r + 8*(ln>>4), n = ln&15)
// ---------------------------------------------------------------------------
__device__ __forceinline__ v16h load_a_rm(const _Float16* base, int ld,
                                          int row0, int k0, int lane) {
  int m = lane & 15, h = lane >> 4;
  const _Float16* p = base + (size_t)(row0 + m) * ld + k0 + h * 8;
  v8h lo = *(const v8h*)p;
  v8h hi = *(const v8h*)(p + 16);
  v16h r;
#pragma unroll
  for (int i = 0; i < 8; ++i) { r[i] = lo[i]; r[i + 8] = hi[i]; }
  return r;
}

// B operand, K-major source: B[k][n] = Bm[(k0+k)*ld + n0+n]
// (lane n = lane&15; elems 0..7 = K[half*8 .. +7], elems 8..15 = K[16+half*8 .. +7])
__device__ __forceinline__ v16h load_b_km(const _Float16* Bm, int ld,
                                          int k0, int n0, int lane) {
  int nn = lane & 15, h = lane >> 4;
  const _Float16* p = Bm + (size_t)(k0 + h * 8) * ld + n0 + nn;
  v16h r;
#pragma unroll
  for (int i = 0; i < 8; ++i) { r[i] = p[(size_t)i * ld]; r[i + 8] = p[(size_t)(16 + i) * ld]; }
  return r;
}

#define WMMA_F16(Aop, Bop, Cacc) \
  __builtin_amdgcn_wmma_f32_16x16x32_f16(false, (Aop), false, (Bop), (short)0, (Cacc), false, false)

// ---------------------------------------------------------------------------
// K1: L2-normalize features -> fn16[T,F] and fnT[F,T];
//     labels (+identity block) -> lab16[T,L] and labT[L,T]; row sums.
//     One wave per row t.
// ---------------------------------------------------------------------------
__global__ void __launch_bounds__(256)
k1_prep(const float* __restrict__ oq, const float* __restrict__ lq,
        const float* __restrict__ proto,
        _Float16* __restrict__ fn, _Float16* __restrict__ fnT,
        _Float16* __restrict__ lab, _Float16* __restrict__ labT,
        float* __restrict__ rowsum) {
  int wave = (blockIdx.x * blockDim.x + threadIdx.x) >> 5;
  int lane = threadIdx.x & 31;
  if (wave >= Tt) return;

  const float* src = (wave < Bq) ? (oq + (size_t)wave * Fd)
                                 : (proto + (size_t)(wave - Bq) * Fd);
  v4f x = *(const v4f*)(src + lane * 4);
  float ss = x[0]*x[0] + x[1]*x[1] + x[2]*x[2] + x[3]*x[3];
#pragma unroll
  for (int o = 16; o >= 1; o >>= 1) ss += __shfl_xor(ss, o, 32);
  float inv = 1.0f / fmaxf(sqrtf(ss), 1e-12f);
  _Float16* dst = fn + (size_t)wave * Fd + lane * 4;
#pragma unroll
  for (int i = 0; i < 4; ++i) {
    _Float16 v = (_Float16)(x[i] * inv);
    dst[i] = v;
    fnT[(size_t)(lane * 4 + i) * Tt + wave] = v;   // transposed copy
  }

  _Float16* lrow = lab + (size_t)wave * Lc;
  float rs = 0.f;
  if (wave < Bq) {
    const float* l = lq + (size_t)wave * Lc + lane * 8;
#pragma unroll
    for (int i = 0; i < 8; ++i) {
      float v = l[i];
      rs += v;
      _Float16 hv = (_Float16)v;
      lrow[lane * 8 + i] = hv;
      labT[(size_t)(lane * 8 + i) * Tt + wave] = hv;
    }
#pragma unroll
    for (int o = 16; o >= 1; o >>= 1) rs += __shfl_xor(rs, o, 32);
  } else {
    int cls = wave - Bq;
#pragma unroll
    for (int i = 0; i < 8; ++i) {
      int c = lane * 8 + i;
      _Float16 hv = (_Float16)((c == cls) ? 1.0f : 0.0f);
      lrow[c] = hv;
      labT[(size_t)c * Tt + wave] = hv;
    }
    rs = 1.0f;
  }
  if (lane == 0) rowsum[wave] = rs;
}

// ---------------------------------------------------------------------------
// K2: q16[a,c] = lab[a,c] / (lab[a,c] * (re_norm @ lab)[a,c] + EPS),
//     also writes qT[c,a]. One wave per 16-row a-tile, loop b in steps of 32:
//     ll strip via WMMA (K=256, B from labT); elementwise -> re_norm f16;
//     C-layout -> A-layout transpose through LDS; 16 WMMAs accumulating
//     re_norm(16x32) x lab(32x256).
// ---------------------------------------------------------------------------
__global__ void __launch_bounds__(128)
k2_qmat(const _Float16* __restrict__ lab, const _Float16* __restrict__ labT,
        const float* __restrict__ rowsum,
        _Float16* __restrict__ q16, _Float16* __restrict__ qT) {
  __shared__ __align__(16) _Float16 smem[4][16 * 32];
  int wv = threadIdx.x >> 5, lane = threadIdx.x & 31;
  int a0 = (blockIdx.x * 4 + wv) * 16;
  int n = lane & 15, h = lane >> 4;
  _Float16* sm = smem[wv];

  v8f acc[16];
  v8f zero = {};
#pragma unroll
  for (int i = 0; i < 16; ++i) acc[i] = zero;

  for (int b0 = 0; b0 < Tt; b0 += 32) {
    // ll strip: two 16x16 tiles, K = Lc = 256
    v8f d1 = zero, d2 = zero;
#pragma unroll
    for (int k = 0; k < Lc; k += 32) {
      v16h Aa = load_a_rm(lab, Lc, a0, k, lane);       // loop-invariant over b0
      v16h B1 = load_b_km(labT, Tt, k, b0, lane);
      v16h B2 = load_b_km(labT, Tt, k, b0 + 16, lane);
      d1 = WMMA_F16(Aa, B1, d1);
      d2 = WMMA_F16(Aa, B2, d2);
    }
    // re_norm = (ll / rs[b])^2 with zero diagonal, converted to f16 in LDS
    float rs1 = rowsum[b0 + n];
    float rs2 = rowsum[b0 + 16 + n];
#pragma unroll
    for (int r = 0; r < 8; ++r) {
      int mm = r + 8 * h;
      int ag = a0 + mm;
      float x1 = d1[r] / rs1; x1 *= x1; if (ag == b0 + n)      x1 = 0.f;
      float x2 = d2[r] / rs2; x2 *= x2; if (ag == b0 + 16 + n) x2 = 0.f;
      sm[mm * 32 + n]      = (_Float16)x1;
      sm[mm * 32 + 16 + n] = (_Float16)x2;
    }
    asm volatile("s_wait_dscnt 0" ::: "memory");  // LDS RAW/WAR within wave
    v16h Ar = load_a_rm(sm, 32, 0, 0, lane);
#pragma unroll
    for (int ct = 0; ct < 16; ++ct) {
      v16h Bc = load_b_km(lab, Lc, b0, ct * 16, lane); // lab is K(b)-major here
      acc[ct] = WMMA_F16(Ar, Bc, acc[ct]);
    }
  }
  // epilogue: q = lab / (lab*acc + EPS), emit row-major and transposed
#pragma unroll
  for (int ct = 0; ct < 16; ++ct) {
#pragma unroll
    for (int r = 0; r < 8; ++r) {
      int ag = a0 + r + 8 * h;
      int c = ct * 16 + n;
      float lv = (float)lab[(size_t)ag * Lc + c];
      float q = lv / (lv * acc[ct][r] + EPSf);
      _Float16 hq = (_Float16)q;
      q16[(size_t)ag * Lc + c] = hq;
      qT[(size_t)c * Tt + ag]  = hq;
    }
  }
}

// ---------------------------------------------------------------------------
// K3: per-row online softmax stats over sim/TEMP (max includes diagonal,
//     denominator excludes it). One wave per 16-row a-tile.
// ---------------------------------------------------------------------------
__global__ void __launch_bounds__(128)
k3_softstats(const _Float16* __restrict__ fn, const _Float16* __restrict__ fnT,
             float* __restrict__ rowmax, float* __restrict__ denom) {
  int wv = threadIdx.x >> 5, lane = threadIdx.x & 31;
  int a0 = (blockIdx.x * 4 + wv) * 16;
  int n = lane & 15, h = lane >> 4;
  const float invT = 1.0f / TEMPf;

  float mrow[8], drow[8];
#pragma unroll
  for (int r = 0; r < 8; ++r) { mrow[r] = -3.0e38f; drow[r] = 0.f; }

  v8f zero = {};
  for (int b0 = 0; b0 < Tt; b0 += 16) {
    v8f d = zero;
#pragma unroll
    for (int k = 0; k < Fd; k += 32) {
      v16h Aa = load_a_rm(fn, Fd, a0, k, lane);
      v16h Bb = load_b_km(fnT, Tt, k, b0, lane);
      d = WMMA_F16(Aa, Bb, d);
    }
#pragma unroll
    for (int r = 0; r < 8; ++r) {
      float mt = d[r] * invT;
      float nm = fmaxf(mrow[r], mt);
      float add = (a0 + r + 8 * h == b0 + n) ? 0.f : __expf(mt - nm);
      drow[r] = drow[r] * __expf(mrow[r] - nm) + add;
      mrow[r] = nm;
    }
  }
  // combine (m, d) pairs across the 16 lanes of each half-group
#pragma unroll
  for (int o = 8; o >= 1; o >>= 1) {
#pragma unroll
    for (int r = 0; r < 8; ++r) {
      float om = __shfl_xor(mrow[r], o, 16);
      float od = __shfl_xor(drow[r], o, 16);
      float nm = fmaxf(mrow[r], om);
      drow[r] = drow[r] * __expf(mrow[r] - nm) + od * __expf(om - nm);
      mrow[r] = nm;
    }
  }
  if (n == 0) {
#pragma unroll
    for (int r = 0; r < 8; ++r) {
      int a = a0 + r + 8 * h;
      rowmax[a] = mrow[r];
      denom[a]  = drow[r];
    }
  }
}

// ---------------------------------------------------------------------------
// K4: fused loss pass. Per tile: sim (K=128, 4 WMMA), ll & p (K=256, share B,
//     8+8 WMMA). w = (ll/rs[b])^2 * p / rs[a]; accumulate
//     A1 = sum w*mt, A2 = sum w, A3 = sum max(sigma - w, 0)*mt over w != 0.
// ---------------------------------------------------------------------------
__global__ void __launch_bounds__(128)
k4_loss(const _Float16* __restrict__ fn, const _Float16* __restrict__ fnT,
        const _Float16* __restrict__ lab, const _Float16* __restrict__ labT,
        const _Float16* __restrict__ q16, const float* __restrict__ rowsum,
        const float* __restrict__ rowmax, const float* __restrict__ denom,
        float* __restrict__ lossrow, float* __restrict__ anchorflag) {
  int wv = threadIdx.x >> 5, lane = threadIdx.x & 31;
  int a0 = (blockIdx.x * 4 + wv) * 16;
  int n = lane & 15, h = lane >> 4;
  const float invT = 1.0f / TEMPf;

  float mrow[8], invd[8], invrsa[8];
#pragma unroll
  for (int r = 0; r < 8; ++r) {
    int a = a0 + r + 8 * h;
    mrow[r]   = rowmax[a];
    invd[r]   = 1.0f / denom[a];
    invrsa[r] = 1.0f / rowsum[a];
  }

  v8f zero = {};
  v8f A1 = zero, A2 = zero, A3 = zero;
  for (int b0 = 0; b0 < Tt; b0 += 16) {
    v8f ds = zero, dl = zero, dp = zero;
#pragma unroll
    for (int k = 0; k < Fd; k += 32) {
      v16h Aa = load_a_rm(fn, Fd, a0, k, lane);
      v16h Bb = load_b_km(fnT, Tt, k, b0, lane);
      ds = WMMA_F16(Aa, Bb, ds);
    }
#pragma unroll
    for (int k = 0; k < Lc; k += 32) {
      v16h Bl = load_b_km(labT, Tt, k, b0, lane);  // shared by ll and p
      v16h Al = load_a_rm(lab, Lc, a0, k, lane);
      v16h Aq = load_a_rm(q16, Lc, a0, k, lane);
      dl = WMMA_F16(Al, Bl, dl);
      dp = WMMA_F16(Aq, Bl, dp);
    }
    float rsb = rowsum[b0 + n];
#pragma unroll
    for (int r = 0; r < 8; ++r) {
      int a = a0 + r + 8 * h;
      if (a == b0 + n) continue;              // diagonal: w==0, reg masked
      float rn = dl[r] / rsb; rn *= rn;       // re_norm (ALPHA == 2)
      float w = rn * dp[r] * invrsa[r];
      if (w != 0.f) {
        float mt = ds[r] * invT;
        A1[r] += w * mt;
        A2[r] += w;
        float sg = __expf(mt - mrow[r]) * invd[r];
        A3[r] += fmaxf(sg - w, 0.f) * mt;
      }
    }
  }
#pragma unroll
  for (int o = 8; o >= 1; o >>= 1) {
#pragma unroll
    for (int r = 0; r < 8; ++r) {
      A1[r] += __shfl_xor(A1[r], o, 16);
      A2[r] += __shfl_xor(A2[r], o, 16);
      A3[r] += __shfl_xor(A3[r], o, 16);
    }
  }
  if (n == 0) {
#pragma unroll
    for (int r = 0; r < 8; ++r) {
      int a = a0 + r + 8 * h;
      float logd = __logf(denom[a]);
      lossrow[a]    = A1[r] - (mrow[r] + logd) * A2[r] + A3[r];
      anchorflag[a] = (A2[r] != 0.f) ? 1.f : 0.f;
    }
  }
}

// ---------------------------------------------------------------------------
// K5: scalar reduction:  out = -sum(lossrow) / sum(anchorflag)
// ---------------------------------------------------------------------------
__global__ void __launch_bounds__(256)
k5_final(const float* __restrict__ lossrow, const float* __restrict__ anchorflag,
         float* __restrict__ out) {
  __shared__ float ssum[256];
  __shared__ float scnt[256];
  float s = 0.f, c = 0.f;
  for (int i = threadIdx.x; i < Tt; i += 256) {
    s += lossrow[i];
    c += anchorflag[i];
  }
  ssum[threadIdx.x] = s;
  scnt[threadIdx.x] = c;
  __syncthreads();
  for (int o = 128; o >= 1; o >>= 1) {
    if ((int)threadIdx.x < o) {
      ssum[threadIdx.x] += ssum[threadIdx.x + o];
      scnt[threadIdx.x] += scnt[threadIdx.x + o];
    }
    __syncthreads();
  }
  if (threadIdx.x == 0) out[0] = -ssum[0] / scnt[0];
}

// ---------------------------------------------------------------------------
extern "C" void kernel_launch(void* const* d_in, const int* in_sizes, int n_in,
                              void* d_out, int out_size, void* d_ws, size_t ws_size,
                              hipStream_t stream) {
  (void)in_sizes; (void)n_in; (void)out_size; (void)ws_size;
  const float* oq    = (const float*)d_in[0];  // [B, F]
  const float* lq    = (const float*)d_in[1];  // [B, L]
  const float* proto = (const float*)d_in[2];  // [L, F]

  char* ws = (char*)d_ws;
  size_t off = 0;
  auto take = [&](size_t bytes) -> char* {
    char* p = ws + off;
    off = (off + bytes + 255) & ~(size_t)255;
    return p;
  };
  _Float16* fn     = (_Float16*)take((size_t)Tt * Fd * sizeof(_Float16)); // 2.2 MB
  _Float16* fnT    = (_Float16*)take((size_t)Fd * Tt * sizeof(_Float16)); // 2.2 MB
  _Float16* lab    = (_Float16*)take((size_t)Tt * Lc * sizeof(_Float16)); // 4.3 MB
  _Float16* labT   = (_Float16*)take((size_t)Lc * Tt * sizeof(_Float16)); // 4.3 MB
  _Float16* q16    = (_Float16*)take((size_t)Tt * Lc * sizeof(_Float16)); // 4.3 MB
  _Float16* qT     = (_Float16*)take((size_t)Lc * Tt * sizeof(_Float16)); // 4.3 MB
  float*    rowsum = (float*)take((size_t)Tt * sizeof(float));
  float*    rowmax = (float*)take((size_t)Tt * sizeof(float));
  float*    denom  = (float*)take((size_t)Tt * sizeof(float));
  float*    lossrow= (float*)take((size_t)Tt * sizeof(float));
  float*    aflag  = (float*)take((size_t)Tt * sizeof(float));
  (void)qT;

  // 8448 rows / 8 waves per 256-thread block
  k1_prep<<<Tt / 8, 256, 0, stream>>>(oq, lq, proto, fn, fnT, lab, labT, rowsum);
  // 528 a-tiles, 4 waves (a-tiles) per 128-thread block
  k2_qmat<<<132, 128, 0, stream>>>(lab, labT, rowsum, q16, qT);
  k3_softstats<<<132, 128, 0, stream>>>(fn, fnT, rowmax, denom);
  k4_loss<<<132, 128, 0, stream>>>(fn, fnT, lab, labT, q16, rowsum,
                                   rowmax, denom, lossrow, aflag);
  k5_final<<<1, 256, 0, stream>>>(lossrow, aflag, (float*)d_out);
}
